// SelfAttention_45672682226115
// MI455X (gfx1250) — matrix operand order repaired and checked by reading, not compiled, not run on hardware
//
#include <hip/hip_runtime.h>

// SAGAN self-attention for MI455X (gfx1250), flash-attention formulation.
// B=4, C=256, H=W=64 -> N=4096, CK=32.
//
//   1) sagan_proj : q/k/v 1x1-conv projections (f32 math, f16 outputs in ws)
//   2) sagan_attn : per-wave 16-query flash attention over 64-key blocks.
//      - K/V tiles double-buffered in LDS via GLOBAL_LOAD_ASYNC_TO_LDS_B128
//        (ASYNCcnt-tracked, no VGPR staging, copy of tile j+1 overlaps
//        compute of tile j).
//      - S^T = K_tile x Q^T via v_wmma_f32_16x16x32_f16 (4 per block), so the
//        softmax reduction is per-lane over VGPRs + one shfl_xor(16).
//      - P^T rebuilt into WMMA B layout with v_cvt_pk_rtz_f16_f32 + bpermute.
//      - O^T += V^T_block x P^T : 32 WMMAs per 64-key block, 128 f32 acc VGPRs.
//
// Workspace: (2*B*N*CK + B*N*C) f16 = 10.5 MB.

#define BB   4
#define CDIM 256
#define NN   4096
#define CKD  32
#define NT   (NN / 64)  // 64-key blocks per batch

typedef _Float16 half_t;
typedef __attribute__((ext_vector_type(16))) _Float16 v16h;
typedef __attribute__((ext_vector_type(2)))  __fp16   fp16x2;
typedef __attribute__((ext_vector_type(8)))  float    v8f;

union V16U { unsigned int u[8]; v16h v; };

__device__ __forceinline__ unsigned int pack_h2(float lo, float hi) {
  union { fp16x2 h; unsigned int u; } c;
  c.h = __builtin_amdgcn_cvt_pkrtz(lo, hi);  // v_cvt_pk_rtz_f16_f32
  return c.u;
}

// Async global->LDS copy of one 16B chunk per lane (wave covers 512B).
__device__ __forceinline__ void async_copy16(const half_t* gbase, unsigned voff,
                                             half_t* lbase, unsigned loff) {
  unsigned lds = (unsigned)(unsigned long long)lbase + loff;
  asm volatile("global_load_async_to_lds_b128 %0, %1, %2"
               :: "v"(lds), "v"(voff),
                  "s"((unsigned long long)gbase)
               : "memory");
}

__device__ __forceinline__ void wait_async0() {
  asm volatile("s_wait_asynccnt 0x0" ::: "memory");
}

// ---------------------------------------------------------------------------
// Projection: o in [0,32)=Wf->Q, [32,64)=Wg->K, [64,320)=Wv->V
// grid (N/256, 320, B), block 256. Weight row is block-uniform -> scalar loads.
// ---------------------------------------------------------------------------
__global__ __launch_bounds__(256) void sagan_proj(
    const float* __restrict__ x,  const float* __restrict__ Wf,
    const float* __restrict__ Wg, const float* __restrict__ Wv,
    half_t* __restrict__ Qh, half_t* __restrict__ Kh, half_t* __restrict__ Vh)
{
  const int n = blockIdx.x * 256 + threadIdx.x;
  const int o = blockIdx.y;
  const int b = blockIdx.z;

  const float* wrow;
  if (o < CKD)          wrow = Wf + (size_t)o * CDIM;
  else if (o < 2 * CKD) wrow = Wg + (size_t)(o - CKD) * CDIM;
  else                  wrow = Wv + (size_t)(o - 2 * CKD) * CDIM;

  const float* xb = x + (size_t)b * CDIM * NN + n;
  float acc = 0.f;
#pragma unroll 8
  for (int c = 0; c < CDIM; ++c)
    acc = fmaf(xb[(size_t)c * NN], wrow[c], acc);

  const half_t hv = (half_t)acc;
  if (o < CKD) {
    Qh[((size_t)b * NN + n) * CKD + o] = hv;
  } else if (o < 2 * CKD) {
    Kh[((size_t)b * NN + n) * CKD + (o - CKD)] = hv;
  } else {
    const int c = o - 2 * CKD;
    Vh[(((size_t)b * (NN / 32) + (n >> 5)) * CDIM + c) * 32 + (n & 31)] = hv;
  }
}

// ---------------------------------------------------------------------------
// Flash attention. grid (N/64, B), block 128 (4 waves). Each wave: 16 queries,
// full C=256 output in 16 x v8f accumulators (O^T layout). 64-key blocks.
// ---------------------------------------------------------------------------
__global__ __launch_bounds__(128) void sagan_attn(
    const float*  __restrict__ x,  const float* __restrict__ gamma,
    const half_t* __restrict__ Qh, const half_t* __restrict__ Kh,
    const half_t* __restrict__ Vh, float* __restrict__ out)
{
  // Double-buffered tiles. K: [buf][key 0..63][ck 0..31]  (4KB/buf)
  //                        V: [buf][h2 0..1][c 0..255][kk 0..31] (32KB/buf)
  __shared__ __align__(16) half_t ldsK[2][64 * CKD];
  __shared__ __align__(16) half_t ldsV[2][64 * CDIM];

  const int b    = blockIdx.y;
  const int tid  = threadIdx.x;
  const int lane = tid & 31;
  const int wv   = tid >> 5;
  const int i0   = blockIdx.x * 64 + wv * 16;  // first query of this wave
  const int col  = lane & 15;
  const int hh   = lane >> 4;

  const half_t* kbat = Kh + (size_t)b * NN * CKD;
  const half_t* vbat = Vh + (size_t)b * NN * CDIM;  // (N/32)*CDIM*32 == N*CDIM

  // Q^T as WMMA B-matrix (32ck x 16 queries).
  V16U qb;
  {
    const half_t* qbase = Qh + ((size_t)b * NN + (i0 + col)) * CKD;
#pragma unroll
    for (int w = 0; w < 8; ++w)
      qb.u[w] = *(const unsigned int*)(qbase + hh * 16 + 2 * w);
  }

  v8f acc[16];
#pragma unroll
  for (int cb = 0; cb < 16; ++cb)
    acc[cb] = (v8f){0.f, 0.f, 0.f, 0.f, 0.f, 0.f, 0.f, 0.f};
  const v8f zero8 = (v8f){0.f, 0.f, 0.f, 0.f, 0.f, 0.f, 0.f, 0.f};

  float m_i = -INFINITY;
  float l_i = 0.f;

  // Prologue: async-stage tile 0 into buffer 0.
  {
    const half_t* kt = kbat;            // tile 0: 64 rows x 32 ck = 4KB
    const half_t* vt = vbat;            // tile 0: 32KB
#pragma unroll
    for (int i = 0; i < 2; ++i)
      async_copy16(kt, (unsigned)((i * 128 + tid) * 16), ldsK[0],
                   (unsigned)((i * 128 + tid) * 16));
#pragma unroll
    for (int i = 0; i < 16; ++i)
      async_copy16(vt, (unsigned)((i * 128 + tid) * 16), ldsV[0],
                   (unsigned)((i * 128 + tid) * 16));
  }

  for (int jt = 0; jt < NT; ++jt) {
    wait_async0();      // my copies of buffer jt&1 are complete
    __syncthreads();    // everyone's are; prior reads of (jt+1)&1 are done

    if (jt + 1 < NT) {  // stage next tile into the other buffer (overlapped)
      const int nb = (jt + 1) & 1;
      const half_t* kt = kbat + (size_t)(jt + 1) * 64 * CKD;
      const half_t* vt = vbat + (size_t)(jt + 1) * 64 * CDIM;
#pragma unroll
      for (int i = 0; i < 2; ++i)
        async_copy16(kt, (unsigned)((i * 128 + tid) * 16), ldsK[nb],
                     (unsigned)((i * 128 + tid) * 16));
#pragma unroll
      for (int i = 0; i < 16; ++i)
        async_copy16(vt, (unsigned)((i * 128 + tid) * 16), ldsV[nb],
                     (unsigned)((i * 128 + tid) * 16));
    }

    const half_t* lk = ldsK[jt & 1];
    const half_t* lv = ldsV[jt & 1];

    // S^T: 4 tiles of (16 keys x 16 queries). A = K rows (16x32, M = key).
    v8f s[4];
#pragma unroll
    for (int t = 0; t < 4; ++t) {
      V16U a;
#pragma unroll
      for (int v = 0; v < 8; ++v) {
        const int kk = hh * 16 + 2 * v;
        a.u[v] = *(const unsigned int*)(lk + (t * 16 + col) * CKD + kk);
      }
      s[t] = __builtin_amdgcn_wmma_f32_16x16x32_f16(false, a.v, false, qb.v,
                                                    (short)0, zero8, false, false);
    }

    // Online softmax over these 64 keys (per query = col).
    float tm = -INFINITY;
#pragma unroll
    for (int t = 0; t < 4; ++t)
#pragma unroll
      for (int v = 0; v < 8; ++v) tm = fmaxf(tm, s[t][v]);
    tm = fmaxf(tm, __shfl_xor(tm, 16, 32));
    const float m_new = fmaxf(m_i, tm);
    const float scale = __expf(m_i - m_new);

    float e[4][8], psum = 0.f;
#pragma unroll
    for (int t = 0; t < 4; ++t)
#pragma unroll
      for (int v = 0; v < 8; ++v) {
        e[t][v] = __expf(s[t][v] - m_new);
        psum += e[t][v];
      }
    psum += __shfl_xor(psum, 16, 32);
    l_i = l_i * scale + psum;
    m_i = m_new;

#pragma unroll
    for (int cb = 0; cb < 16; ++cb)
#pragma unroll
      for (int v = 0; v < 8; ++v) acc[cb][v] *= scale;

    // Rebuild P^T into WMMA B layout. For key half h2 (32 keys), B VGPR w
    // wants keys {h2*32 + hh*16 + 2w, +1} for query col -> source tile
    // t = 2*h2 + hh, packed pair pk[t][w&3], source lane ((w&4)<<2)+col.
    unsigned int pk[4][4];
#pragma unroll
    for (int t = 0; t < 4; ++t)
#pragma unroll
      for (int s2 = 0; s2 < 4; ++s2)
        pk[t][s2] = pack_h2(e[t][2 * s2], e[t][2 * s2 + 1]);

    V16U pb[2];
#pragma unroll
    for (int h2 = 0; h2 < 2; ++h2)
#pragma unroll
      for (int w = 0; w < 8; ++w) {
        const int src = ((w & 4) << 2) + col;
        const unsigned int blo = (unsigned int)__shfl((int)pk[2 * h2][w & 3], src, 32);
        const unsigned int bhi = (unsigned int)__shfl((int)pk[2 * h2 + 1][w & 3], src, 32);
        pb[h2].u[w] = (hh == 0) ? blo : bhi;
      }

    // O^T += V^T_block (16 channels x 32 keys) x P^T, both key halves.
#pragma unroll
    for (int cb = 0; cb < 16; ++cb) {
      V16U av0, av1;
#pragma unroll
      for (int v = 0; v < 8; ++v) {
        const int kk = hh * 16 + 2 * v;
        av0.u[v] = *(const unsigned int*)(lv + (cb * 16 + col) * 32 + kk);
        av1.u[v] = *(const unsigned int*)(lv + 8192 + (cb * 16 + col) * 32 + kk);
      }
      acc[cb] = __builtin_amdgcn_wmma_f32_16x16x32_f16(false, av0.v, false, pb[0].v,
                                                       (short)0, acc[cb], false, false);
      acc[cb] = __builtin_amdgcn_wmma_f32_16x16x32_f16(false, av1.v, false, pb[1].v,
                                                       (short)0, acc[cb], false, false);
    }
  }

  // Epilogue: out = gamma * O / l + x.
  const float g  = gamma[0];
  const float rl = 1.0f / l_i;
#pragma unroll
  for (int cb = 0; cb < 16; ++cb) {
#pragma unroll
    for (int v = 0; v < 8; ++v) {
      const int c = cb * 16 + v + hh * 8;
      const size_t idx = ((size_t)b * CDIM + c) * NN + (size_t)(i0 + col);
      out[idx] = fmaf(g, acc[cb][v] * rl, x[idx]);
    }
  }
}

// ---------------------------------------------------------------------------
extern "C" void kernel_launch(void* const* d_in, const int* in_sizes, int n_in,
                              void* d_out, int out_size, void* d_ws, size_t ws_size,
                              hipStream_t stream) {
  const float* x     = (const float*)d_in[0];
  const float* Wf    = (const float*)d_in[1];
  const float* Wg    = (const float*)d_in[2];
  const float* Wv    = (const float*)d_in[3];
  const float* gamma = (const float*)d_in[4];
  float*       out   = (float*)d_out;

  // Workspace layout (f16): Q | K | V  -> 1MB + 1MB + 8MB
  half_t* Qh = (half_t*)d_ws;
  half_t* Kh = Qh + (size_t)BB * NN * CKD;
  half_t* Vh = Kh + (size_t)BB * NN * CKD;

  sagan_proj<<<dim3(NN / 256, 2 * CKD + CDIM, BB), 256, 0, stream>>>(
      x, Wf, Wg, Wv, Qh, Kh, Vh);
  sagan_attn<<<dim3(NN / 64, BB), 128, 0, stream>>>(
      x, gamma, Qh, Kh, Vh, out);
}
